// VanillaAttention_46823733461301
// MI455X (gfx1250) — matrix operand order repaired and checked by reading, not compiled, not run on hardware
//
#include <hip/hip_runtime.h>
#include <hip/hip_bf16.h>

// Flash-attention (causal) for B=4, L=2048, H=16, E=64, fp32 in/out.
// f16 WMMA (v_wmma_f32_16x16x32_f16) with fp32 accumulation.
// K/V tiles double-buffered in LDS via GLOBAL_LOAD_ASYNC_TO_LDS_B128
// (ASYNCcnt-tracked) when the toolchain exposes the builtins.

typedef __attribute__((ext_vector_type(16))) _Float16 v16h;
typedef __attribute__((ext_vector_type(8)))  float    v8f;

union AFrag { v16h v; _Float16 h[16]; };
union CFrag { v8f  v; float     f[8];  };
union F8    { v8f  v; float     f[8];  };

#define FA_B 4
#define FA_L 2048
#define FA_H 16
#define FA_E 64
#define FA_QT 64   // query rows per block
#define FA_KT 32   // keys per inner iteration
#define FA_NW 4    // waves per block

// 0.125 (1/sqrt(64)) folded with log2(e) so exp2f is used directly.
#define FA_SCL 0.18033688011112042f

#if defined(__HIP_DEVICE_COMPILE__) && \
    __has_builtin(__builtin_amdgcn_global_load_async_to_lds_b128) && \
    __has_builtin(__builtin_amdgcn_s_wait_asynccnt)
#define FA_ASYNC 1
// Builtin expects pointers to int vector_size(16): global AS(1) src, LDS AS(3) dst.
typedef int v4i_b128 __attribute__((vector_size(16)));
typedef __attribute__((address_space(1))) v4i_b128 gv4i;
typedef __attribute__((address_space(3))) v4i_b128 lv4i;
#else
#define FA_ASYNC 0
#endif

__global__ __launch_bounds__(128) void fa_causal_wmma_kernel(
    const float* __restrict__ qkv, float* __restrict__ out)
{
    const int tid  = threadIdx.x;
    const int wave = tid >> 5;
    const int lane = tid & 31;
    const int hi   = lane >> 4;     // lane-group (0: lanes 0-15, 1: lanes 16-31)
    const int n    = lane & 15;     // column index within C/B layouts, row for A
    const int m    = n;             // A-fragment row

    const int nQB = FA_L / FA_QT;                       // 32
    const int qbi = blockIdx.x % nQB;
    const int h   = (blockIdx.x / nQB) % FA_H;
    const int b   = blockIdx.x / (nQB * FA_H);
    const int qb  = qbi * FA_QT;

    // LDS: raw f32 K/V tiles (double buffered, async targets),
    // f16 transposed V for PV B-operands, per-wave P scratch.
    __shared__ float    rawK[2][FA_KT * FA_E];          // 16 KB
    __shared__ float    rawV[2][FA_KT * FA_E];          // 16 KB
    __shared__ _Float16 sVt [FA_E * FA_KT];             // 4 KB
    __shared__ _Float16 sP  [FA_NW][16 * FA_KT];        // 4 KB

    // Per-thread slice of a K/V tile copy: key row = tid/4, 16-float chunk.
    const int cKey = tid >> 2;
    const int cE0  = (tid & 3) * 16;

    // ---- Tile fetch: global (f32) -> raw LDS, async when available ----
    auto issue_copy = [&](int kt, int buf) {
        const long kg = (long)kt * FA_KT + cKey;
        const float* kp = qkv + ((kg + (long)b * FA_L) * 3 + 1) * (FA_H * FA_E) + h * FA_E + cE0;
        const float* vp = qkv + ((kg + (long)b * FA_L) * 3 + 2) * (FA_H * FA_E) + h * FA_E + cE0;
        float* dK = &rawK[buf][cKey * FA_E + cE0];
        float* dV = &rawV[buf][cKey * FA_E + cE0];
#if FA_ASYNC
        // 8 async b128 copies per thread per tile (4 for K, 4 for V).
        __builtin_amdgcn_global_load_async_to_lds_b128((gv4i*)kp, (lv4i*)dK,  0, 0);
        __builtin_amdgcn_global_load_async_to_lds_b128((gv4i*)kp, (lv4i*)dK, 16, 0);
        __builtin_amdgcn_global_load_async_to_lds_b128((gv4i*)kp, (lv4i*)dK, 32, 0);
        __builtin_amdgcn_global_load_async_to_lds_b128((gv4i*)kp, (lv4i*)dK, 48, 0);
        __builtin_amdgcn_global_load_async_to_lds_b128((gv4i*)vp, (lv4i*)dV,  0, 0);
        __builtin_amdgcn_global_load_async_to_lds_b128((gv4i*)vp, (lv4i*)dV, 16, 0);
        __builtin_amdgcn_global_load_async_to_lds_b128((gv4i*)vp, (lv4i*)dV, 32, 0);
        __builtin_amdgcn_global_load_async_to_lds_b128((gv4i*)vp, (lv4i*)dV, 48, 0);
#else
#pragma unroll
        for (int i = 0; i < 16; ++i) { dK[i] = kp[i]; dV[i] = vp[i]; }
#endif
    };

    // ---- Load Q fragments for this wave's 16 rows (f32 -> f16) ----
    const long rowQ = (long)qb + wave * 16 + m;
    const float* qp = qkv + ((rowQ + (long)b * FA_L) * 3 + 0) * (FA_H * FA_E) + h * FA_E;

    AFrag qa[2];
#pragma unroll
    for (int f = 0; f < 2; ++f) {
        const int c0 = f * 32 + hi * 16;       // slots 0..7
        const int c1 = f * 32 + hi * 16 + 8;   // slots 8..15
#pragma unroll
        for (int i = 0; i < 8; ++i) {
            qa[f].h[i]     = (_Float16)qp[c0 + i];
            qa[f].h[8 + i] = (_Float16)qp[c1 + i];
        }
    }

    // ---- Online softmax state ----
    CFrag acc[4];
#pragma unroll
    for (int t = 0; t < 4; ++t) acc[t].v = (v8f){};
    float rm[8], rl[8];
#pragma unroll
    for (int r = 0; r < 8; ++r) { rm[r] = -__builtin_inff(); rl[r] = 0.0f; }

    const int ktmax = qb / FA_KT + 2;   // causal: keys < qb + 64 (>= 2 always)

    issue_copy(0, 0);                   // prologue prefetch

    for (int kt = 0; kt < ktmax; ++kt) {
        const int cur = kt & 1;

        __syncthreads();                // all compute on buf cur^1 / sVt done
        if (kt + 1 < ktmax) {
            issue_copy(kt + 1, cur ^ 1);
#if FA_ASYNC
            __builtin_amdgcn_s_wait_asynccnt(8);   // tile kt's copies landed
#endif
        } else {
#if FA_ASYNC
            __builtin_amdgcn_s_wait_asynccnt(0);
#endif
        }
        __syncthreads();                // tile kt raw data visible block-wide

        // ---- V transpose/convert: rawV[cur] [key][e] f32 -> sVt [e][key] f16 ----
        {
            const float* src = &rawV[cur][cKey * FA_E + cE0];
#pragma unroll
            for (int i = 0; i < 16; ++i)
                sVt[(cE0 + i) * FA_KT + cKey] = (_Float16)src[i];
        }
        __syncthreads();                // sVt ready

        // Wave active if any key in this tile is <= its max row index.
        const bool active = (kt * FA_KT) <= (qb + wave * 16 + 15);
        if (active) {
            // ---- S = Q K^T for two 16-key score tiles ----
            CFrag s[2];
#pragma unroll
            for (int st = 0; st < 2; ++st) {
                const int keyb = st * 16 + n;   // this lane's key column
                // B-operand from raw f32 K: 16 contiguous floats per lane, cvt to f16.
                const float* kr = &rawK[cur][keyb * FA_E + hi * 16];
                F8 x0, x1, x2, x3;
                x0.v = *(const v8f*)(kr);       x1.v = *(const v8f*)(kr + 8);
                x2.v = *(const v8f*)(kr + 32);  x3.v = *(const v8f*)(kr + 40);
                AFrag kb0, kb1;
#pragma unroll
                for (int i = 0; i < 8; ++i) {
                    kb0.h[i]     = (_Float16)x0.f[i];
                    kb0.h[8 + i] = (_Float16)x1.f[i];
                    kb1.h[i]     = (_Float16)x2.f[i];
                    kb1.h[8 + i] = (_Float16)x3.f[i];
                }
                v8f c = {};
                c = __builtin_amdgcn_wmma_f32_16x16x32_f16(false, qa[0].v, false, kb0.v,
                                                           (short)0, c, false, false);
                c = __builtin_amdgcn_wmma_f32_16x16x32_f16(false, qa[1].v, false, kb1.v,
                                                           (short)0, c, false, false);
                s[st].v = c;
            }

            // ---- scale (folded with log2 e) + causal mask ----
#pragma unroll
            for (int st = 0; st < 2; ++st) {
#pragma unroll
                for (int r = 0; r < 8; ++r) {
                    const int Mg = qb + wave * 16 + r + 8 * hi;
                    const int Ng = kt * FA_KT + st * 16 + n;
                    float x = s[st].f[r] * FA_SCL;
                    s[st].f[r] = (Ng > Mg) ? -__builtin_inff() : x;
                }
            }

            // ---- online softmax in base-2 domain (reduce across 16 lanes) ----
            float alpha[8];
#pragma unroll
            for (int r = 0; r < 8; ++r) {
                float mx = fmaxf(s[0].f[r], s[1].f[r]);
                mx = fmaxf(mx, __shfl_xor(mx, 1, 32));
                mx = fmaxf(mx, __shfl_xor(mx, 2, 32));
                mx = fmaxf(mx, __shfl_xor(mx, 4, 32));
                mx = fmaxf(mx, __shfl_xor(mx, 8, 32));
                const float mnew = fmaxf(rm[r], mx);
                alpha[r] = exp2f(rm[r] - mnew);
                rm[r] = mnew;
                const float p0 = exp2f(s[0].f[r] - mnew);
                const float p1 = exp2f(s[1].f[r] - mnew);
                s[0].f[r] = p0; s[1].f[r] = p1;
                float sum = p0 + p1;
                sum += __shfl_xor(sum, 1, 32);
                sum += __shfl_xor(sum, 2, 32);
                sum += __shfl_xor(sum, 4, 32);
                sum += __shfl_xor(sum, 8, 32);
                rl[r] = rl[r] * alpha[r] + sum;
            }

            // ---- rescale accumulators ----
#pragma unroll
            for (int t = 0; t < 4; ++t)
#pragma unroll
                for (int r = 0; r < 8; ++r) acc[t].f[r] *= alpha[r];

            // ---- P: C layout -> LDS -> A layout (same-wave DS, in order) ----
            _Float16* P = sP[wave];
#pragma unroll
            for (int st = 0; st < 2; ++st)
#pragma unroll
                for (int r = 0; r < 8; ++r)
                    P[(r + 8 * hi) * FA_KT + st * 16 + n] = (_Float16)s[st].f[r];

            AFrag pa;
#pragma unroll
            for (int i = 0; i < 8; ++i) {
                pa.h[i]     = P[m * FA_KT + hi * 16 + i];
                pa.h[8 + i] = P[m * FA_KT + hi * 16 + 8 + i];
            }

            // ---- O += P V : 4 output column tiles ----
#pragma unroll
            for (int t = 0; t < 4; ++t) {
                const v16h* vb = (const v16h*)&sVt[(t * 16 + n) * FA_KT + hi * 16];
                acc[t].v = __builtin_amdgcn_wmma_f32_16x16x32_f16(false, pa.v, false, *vb,
                                                                  (short)0, acc[t].v,
                                                                  false, false);
            }
        }
    }

    // ---- Epilogue: normalize by row sum, store fp32 out[b][l][h][e] ----
#pragma unroll
    for (int r = 0; r < 8; ++r) {
        const float inv = 1.0f / rl[r];
        const long rowg = (long)qb + wave * 16 + r + 8 * hi;
        float* op = out + (((long)b * FA_L + rowg) * FA_H + h) * FA_E;
#pragma unroll
        for (int t = 0; t < 4; ++t)
            op[t * 16 + n] = acc[t].f[r] * inv;
    }
}

extern "C" void kernel_launch(void* const* d_in, const int* in_sizes, int n_in,
                              void* d_out, int out_size, void* d_ws, size_t ws_size,
                              hipStream_t stream) {
    (void)in_sizes; (void)n_in; (void)out_size; (void)d_ws; (void)ws_size;
    const float* qkv = (const float*)d_in[0];
    float* out = (float*)d_out;
    const int nBlocks = FA_B * FA_H * (FA_L / FA_QT);   // 4*16*32 = 2048
    fa_causal_wmma_kernel<<<nBlocks, 128, 0, stream>>>(qkv, out);
}
